// GatedAttentionUnit_54554674594256
// MI455X (gfx1250) — compile-verified, matrix operand order checked by reading, not run on hardware
//
#include <hip/hip_runtime.h>
#include <hip/hip_bf16.h>
#include <math.h>
#include <stdint.h>

typedef __attribute__((ext_vector_type(16))) _Float16 v16h;
typedef __attribute__((ext_vector_type(8)))  float    v8f;
typedef __attribute__((ext_vector_type(4)))  unsigned int uint32x4;
typedef __attribute__((ext_vector_type(4)))  int          int32x4;
typedef __attribute__((ext_vector_type(8)))  int          int32x8;

#define HIDDEN 768
#define KEY    128
#define BATCH  8
#define SEQ    2048
#define INW    384          // HIDDEN/2
#define UVQK   1792         // 2*HIDDEN + 2*KEY
#define QSCALE 0.08838834764831845f   // KEY^-0.5

#define HAS_TDM __has_builtin(__builtin_amdgcn_tensor_load_to_lds)

#if HAS_TDM
#pragma message("CDNA5: tensor_load_to_lds builtin AVAILABLE - TDM path enabled")
#if __has_include(<hip/amd_detail/amd_gfx1250_TDM.h>)
#pragma message("CDNA5: using 6-arg TDM builtin (therock headers present)")
#else
#pragma message("CDNA5: using 5-arg TDM builtin (ROCm 7.2)")
#endif
#else
#pragma message("CDNA5: tensor_load_to_lds builtin NOT available - cooperative-copy fallback")
#endif

// Map element index i (0..15) of a v16h WMMA A/B fragment to its K index,
// per the CDNA5 ISA 16-bit A-matrix 16x32 layout (B assumed symmetric):
//   VGPR0: K=(0,1)|(8,9)  VGPR1: K=(2,3)|(10,11) ... VGPR4..7: +16
// hi = (lane >= 16) selects the second K sub-range.
__device__ __forceinline__ int kmap(int i, int hi) {
    int v = i >> 1, h = i & 1;
    return ((v & 4) ? 16 : 0) + ((v & 3) << 1) + (hi ? 8 : 0) + h;
}

__device__ __forceinline__ float silu_f(float x) {
    return x / (1.0f + __expf(-x));
}

#if HAS_TDM
// Issue one TDM 2D tile load: global (row-major) -> LDS.
// dim0 = elements per row, rows = #rows, stride0 = row pitch (elements),
// ds_code: 0=1B 1=2B 2=4B.  Tensor dims == tile dims (no OOB), type=2.
__device__ __forceinline__ void tdm_load_2d(const void* gptr, unsigned lds_off,
                                            unsigned dim0, unsigned rows,
                                            unsigned stride0, unsigned ds_code) {
    unsigned long long ga = (unsigned long long)(uintptr_t)gptr;
    uint32x4 g0;
    g0[0] = 1u;                                          // count=1 user D#
    g0[1] = lds_off;                                     // lds byte address
    g0[2] = (unsigned)(ga & 0xFFFFFFFFu);                // global_addr[31:0]
    g0[3] = (unsigned)((ga >> 32) & 0x1FFFFFFu) | (2u << 30);  // [56:32] + type=2
    int32x8 g1;
    g1[0] = (int)(ds_code << 16);                        // data_size
    g1[1] = (int)((dim0 & 0xFFFFu) << 16);               // tensor_dim0 lo
    g1[2] = (int)((dim0 >> 16) | ((rows & 0xFFFFu) << 16)); // dim0 hi | dim1 lo
    g1[3] = (int)((rows >> 16) | ((dim0 & 0xFFFFu) << 16)); // dim1 hi | tile_dim0
    g1[4] = (int)(rows & 0xFFFFu);                       // tile_dim1 (tile_dim2=0)
    g1[5] = (int)stride0;                                // tensor_dim0_stride lo
    g1[6] = 0;
    g1[7] = 0;
    int32x4 z4 = {0, 0, 0, 0};
#if __has_include(<hip/amd_detail/amd_gfx1250_TDM.h>)
    int32x8 z8 = {0, 0, 0, 0, 0, 0, 0, 0};
    __builtin_amdgcn_tensor_load_to_lds(g0, g1, z4, z4, z8, 0);   // clang-23 arity
#else
    __builtin_amdgcn_tensor_load_to_lds(g0, g1, z4, z4, 0);       // ROCm 7.2 arity
#endif
}
#endif

// ---------------------------------------------------------------------------
// Kernel 1: uvqk = x @ W_uv for a 16-row x 128-col stripe, then epilogue:
//   cols [0,768) -> silu -> U ; [768,1536) -> silu -> V ;
//   [1536,1664) -> RoPE -> Q ; [1664,1792) -> RoPE -> K   (f16 outputs)
// W_uv B-panels (32x128 f32, 16KB) are TDM double-buffered through LDS.
// ---------------------------------------------------------------------------
__global__ void __launch_bounds__(256)
proj_kernel(const float* __restrict__ x, const float* __restrict__ pos,
            const float* __restrict__ Wuv,
            _Float16* __restrict__ U, _Float16* __restrict__ V,
            _Float16* __restrict__ Q, _Float16* __restrict__ Kb) {
    __shared__ float sT[16][128];
    __shared__ float sW[2][32 * 128];     // double-buffered W_uv panel

    const int rt     = blockIdx.x;        // 16-row tile id over B*L
    const int stripe = blockIdx.y;        // 0..13 (128 cols each)
    const int bb     = rt >> 7;
    const int l0     = (rt & 127) << 4;
    const int wave   = threadIdx.x >> 5;
    const int lane   = threadIdx.x & 31;
    const int n      = lane & 15;
    const int hi     = lane >> 4;

#if HAS_TDM
    if (wave == 0)                        // prologue: panel for kc=0
        tdm_load_2d(&Wuv[(size_t)0 * UVQK + stripe * 128],
                    (unsigned)(uintptr_t)&sW[0][0], 128, 32, UVQK, 2);
#endif

    v8f acc = {};
    for (int it = 0; it < INW / 32; ++it) {
        const int kc  = it * 32;
        const int cur = it & 1;
#if HAS_TDM
        if (wave == 0) {
            if (it + 1 < INW / 32) {      // prefetch next panel
                tdm_load_2d(&Wuv[(size_t)(kc + 32) * UVQK + stripe * 128],
                            (unsigned)(uintptr_t)&sW[1 - cur][0], 128, 32, UVQK, 2);
                __builtin_amdgcn_s_wait_tensorcnt(1);
            } else {
                __builtin_amdgcn_s_wait_tensorcnt(0);
            }
        }
#else
        for (int idx = threadIdx.x; idx < 32 * 128 / 4; idx += 256) {
            int r = idx >> 5, c = idx & 31;   // 32 uint4 per 128-float row
            ((uint4*)&sW[cur][r * 128])[c] =
                ((const uint4*)&Wuv[(size_t)(kc + r) * UVQK + stripe * 128])[c];
        }
#endif
        __syncthreads();

        if (kc + 32 < INW)   // pull next A rows toward L2 (global_prefetch_b8)
            __builtin_prefetch(&x[(size_t)(bb * SEQ + l0 + n) * INW + kc + 32], 0, 1);
        v16h a, b;
#pragma unroll
        for (int i = 0; i < 16; ++i) {
            int kk = kmap(i, hi);
            a[i] = (_Float16)x[(size_t)(bb * SEQ + l0 + n) * INW + kc + kk];
            b[i] = (_Float16)sW[cur][kk * 128 + wave * 16 + n];
        }
        acc = __builtin_amdgcn_wmma_f32_16x16x32_f16(false, a, false, b,
                                                     (short)0, acc, false, false);
        __syncthreads();                  // panel consumed before next fill
    }
#pragma unroll
    for (int r = 0; r < 8; ++r)
        sT[r + hi * 8][wave * 16 + n] = acc[r];
    __syncthreads();

    const int e0 = threadIdx.x * 8;       // 2048 elems / 256 threads
    for (int e = e0; e < e0 + 8; ++e) {
        int m = e >> 7;
        int c = e & 127;
        int col = stripe * 128 + c;
        int l = l0 + m;
        size_t rowbase = (size_t)(bb * SEQ + l);
        float val = sT[m][c];
        if (col < HIDDEN) {
            U[rowbase * HIDDEN + col] = (_Float16)silu_f(val);
        } else if (col < 2 * HIDDEN) {
            V[rowbase * HIDDEN + (col - HIDDEN)] = (_Float16)silu_f(val);
        } else {
            float out;
            if (c < 64) {
                float sn = pos[(size_t)l * 256 + c];
                float cs = pos[(size_t)l * 256 + 64 + c];
                out = val * cs - sT[m][c + 64] * sn;
            } else {
                int d = c - 64;
                float sn = pos[(size_t)l * 256 + d];
                float cs = pos[(size_t)l * 256 + 64 + d];
                out = val * cs + sT[m][c - 64] * sn;
            }
            if (col < 2 * HIDDEN + KEY)
                Q[rowbase * KEY + c] = (_Float16)out;
            else
                Kb[rowbase * KEY + c] = (_Float16)out;
        }
    }
}

// ---------------------------------------------------------------------------
// Kernel 2: flash attention per 16-query tile, TDM-staged and double-buffered.
//   K tile (32x128 f16, 8KB) + V tile (32x768 f16, 48KB) per key block,
//   streamed global->LDS while the previous block is consumed.
//   Waves 0-1 compute S = Q Kj^T; threads 0..15 run the online softmax;
//   all 8 waves rescale their 16x96 fp32 accumulators and do P @ Vj.
//   Final: acc /= l, gate with U, store G (f16).
// ---------------------------------------------------------------------------
#define LDS_KBYTES (32 * KEY * 2)                 // 8192
#define LDS_BUF    (LDS_KBYTES + 32 * HIDDEN * 2) // 57344
#define LDS_S      (2 * LDS_BUF)                  // 114688
#define LDS_P      (LDS_S + 16 * 32 * 4)          // 116736
#define LDS_R      (LDS_P + 16 * 32 * 2)          // 117760
#define LDS_TOTAL  (LDS_R + 64)                   // 117824

__global__ void __launch_bounds__(256)
attn_kernel(const _Float16* __restrict__ Q, const _Float16* __restrict__ Kb,
            const _Float16* __restrict__ V, const _Float16* __restrict__ U,
            _Float16* __restrict__ G) {
    extern __shared__ char smem[];
    float*    sS = (float*)(smem + LDS_S);
    _Float16* sP = (_Float16*)(smem + LDS_P);
    float*    sR = (float*)(smem + LDS_R);

    const int rt   = blockIdx.x;
    const int bb   = rt >> 7;
    const int l0   = (rt & 127) << 4;
    const int wave = threadIdx.x >> 5;
    const int lane = threadIdx.x & 31;
    const int n    = lane & 15;
    const int hi   = lane >> 4;

    v8f acc[6];
#pragma unroll
    for (int t = 0; t < 6; ++t) { v8f z = {}; acc[t] = z; }

    v16h qf[4];
    if (wave < 2) {             // waves 0 and 1 each own one S tile
#pragma unroll
        for (int dc = 0; dc < 4; ++dc)
#pragma unroll
            for (int i = 0; i < 16; ++i)
                qf[dc][i] = Q[(size_t)(bb * SEQ + l0 + n) * KEY + dc * 32 + kmap(i, hi)];
    }
    float mrow = -1e30f, lrow = 0.0f;

#if HAS_TDM
    const unsigned lds_base = (unsigned)(uintptr_t)smem;  // LDS byte offset
    if (wave == 0) {            // prologue: stream key block 0 into buffer 0
        tdm_load_2d(&Kb[(size_t)(bb * SEQ) * KEY],    lds_base,              KEY,    32, KEY,    1);
        tdm_load_2d(&V [(size_t)(bb * SEQ) * HIDDEN], lds_base + LDS_KBYTES, HIDDEN, 32, HIDDEN, 1);
    }
#endif

    for (int jb = 0; jb < SEQ / 32; ++jb) {
        const int cur = jb & 1;
        const int j   = jb * 32;
        const _Float16* sK = (const _Float16*)(smem + cur * LDS_BUF);
        const _Float16* sV = (const _Float16*)(smem + cur * LDS_BUF + LDS_KBYTES);

#if HAS_TDM
        if (wave == 0) {
            if (jb + 1 < SEQ / 32) {   // prefetch next block into other buffer
                const int nj = j + 32;
                const unsigned nb = lds_base + (1 - cur) * LDS_BUF;
                tdm_load_2d(&Kb[(size_t)(bb * SEQ + nj) * KEY],    nb,              KEY,    32, KEY,    1);
                tdm_load_2d(&V [(size_t)(bb * SEQ + nj) * HIDDEN], nb + LDS_KBYTES, HIDDEN, 32, HIDDEN, 1);
                __builtin_amdgcn_s_wait_tensorcnt(2);   // current pair landed
            } else {
                __builtin_amdgcn_s_wait_tensorcnt(0);   // drain on last block
            }
        }
#else
        {   // cooperative fallback copy: 56KB in 16B chunks, 256 threads
            const uint4* gk = (const uint4*)&Kb[(size_t)(bb * SEQ + j) * KEY];
            const uint4* gv = (const uint4*)&V [(size_t)(bb * SEQ + j) * HIDDEN];
            uint4* lk = (uint4*)sK;
            uint4* lv = (uint4*)sV;
            for (int c = threadIdx.x; c < LDS_KBYTES / 16; c += 256) lk[c] = gk[c];
            for (int c = threadIdx.x; c < 32 * HIDDEN * 2 / 16; c += 256) lv[c] = gv[c];
        }
#endif
        __syncthreads();

        if (wave < 2) {            // S tile t2 = wave, 16x16 each
            const int t2 = wave;
            v8f s = {};
#pragma unroll
            for (int dc = 0; dc < 4; ++dc) {
                v16h bf;
#pragma unroll
                for (int i = 0; i < 16; ++i)
                    bf[i] = sK[(t2 * 16 + n) * KEY + dc * 32 + kmap(i, hi)];
                s = __builtin_amdgcn_wmma_f32_16x16x32_f16(false, qf[dc], false, bf,
                                                           (short)0, s, false, false);
            }
#pragma unroll
            for (int r = 0; r < 8; ++r)
                sS[(r + hi * 8) * 32 + t2 * 16 + n] = s[r];
        }
        __syncthreads();

        if (threadIdx.x < 16) {    // online softmax, one thread per query row
            int row = threadIdx.x;
            float sv[32];
            float mnew = mrow;
#pragma unroll
            for (int c = 0; c < 32; ++c) {
                sv[c] = sS[row * 32 + c] * QSCALE;
                mnew = fmaxf(mnew, sv[c]);
            }
            float r = __expf(mrow - mnew);
            float lsum = 0.0f;
#pragma unroll
            for (int c = 0; c < 32; ++c) {
                float p = __expf(sv[c] - mnew);
                lsum += p;
                sP[row * 32 + c] = (_Float16)p;
            }
            lrow = lrow * r + lsum;
            mrow = mnew;
            sR[row] = r;
        }
        __syncthreads();

        v16h pf;
#pragma unroll
        for (int i = 0; i < 16; ++i)
            pf[i] = sP[n * 32 + kmap(i, hi)];
#pragma unroll
        for (int t = 0; t < 6; ++t) {
#pragma unroll
            for (int r = 0; r < 8; ++r)
                acc[t][r] *= sR[r + hi * 8];
            v16h bf;
            int col = wave * 96 + t * 16 + n;
#pragma unroll
            for (int i = 0; i < 16; ++i)
                bf[i] = sV[kmap(i, hi) * HIDDEN + col];
            acc[t] = __builtin_amdgcn_wmma_f32_16x16x32_f16(false, pf, false, bf,
                                                            (short)0, acc[t], false, false);
        }
        __syncthreads();           // protect LDS buffers before next fill
    }

    if (threadIdx.x < 16) sR[threadIdx.x] = 1.0f / lrow;
    __syncthreads();

#pragma unroll
    for (int t = 0; t < 6; ++t) {
        int col = wave * 96 + t * 16 + n;
#pragma unroll
        for (int r = 0; r < 8; ++r) {
            int m = r + hi * 8;
            size_t rowbase = (size_t)(bb * SEQ + l0 + m);
            float g = acc[t][r] * sR[m] * (float)U[rowbase * HIDDEN + col];
            G[rowbase * HIDDEN + col] = (_Float16)g;
        }
    }
}

// ---------------------------------------------------------------------------
// Kernel 3: out = G @ W_out.  256-thread blocks: the 16x768 f16 G row-panel
// (24KB) is TDM'd into LDS once and shared by all 8 waves; each wave produces
// 3 of the 24 16x16 output tiles (K = 768).
// ---------------------------------------------------------------------------
__global__ void __launch_bounds__(256)
outgemm_kernel(const _Float16* __restrict__ G, const float* __restrict__ Wout,
               float* __restrict__ out) {
    __shared__ _Float16 sG[16 * HIDDEN];  // 24KB

    const int rt   = blockIdx.x;
    const int bb   = rt >> 7;
    const int l0   = (rt & 127) << 4;
    const int wave = threadIdx.x >> 5;
    const int lane = threadIdx.x & 31;
    const int n    = lane & 15;
    const int hi   = lane >> 4;

#if HAS_TDM
    if (wave == 0) {
        tdm_load_2d(&G[(size_t)(bb * SEQ + l0) * HIDDEN],
                    (unsigned)(uintptr_t)sG, HIDDEN, 16, HIDDEN, 1);
        __builtin_amdgcn_s_wait_tensorcnt(0);
    }
#else
    {
        const uint4* gg = (const uint4*)&G[(size_t)(bb * SEQ + l0) * HIDDEN];
        uint4* lg = (uint4*)sG;
        for (int c = threadIdx.x; c < 16 * HIDDEN * 2 / 16; c += 256) lg[c] = gg[c];
    }
#endif
    __syncthreads();

    v8f acc[3];
#pragma unroll
    for (int t = 0; t < 3; ++t) { v8f z = {}; acc[t] = z; }

    for (int kc = 0; kc < HIDDEN; kc += 32) {
        v16h a;
#pragma unroll
        for (int i = 0; i < 16; ++i)
            a[i] = sG[n * HIDDEN + kc + kmap(i, hi)];
#pragma unroll
        for (int tt = 0; tt < 3; ++tt) {
            const int ct = wave + tt * 8;       // 24 col tiles over 8 waves
            v16h b;
#pragma unroll
            for (int i = 0; i < 16; ++i)
                b[i] = (_Float16)Wout[(size_t)(kc + kmap(i, hi)) * INW + ct * 16 + n];
            acc[tt] = __builtin_amdgcn_wmma_f32_16x16x32_f16(false, a, false, b,
                                                             (short)0, acc[tt], false, false);
        }
    }
#pragma unroll
    for (int tt = 0; tt < 3; ++tt) {
        const int ct = wave + tt * 8;
#pragma unroll
        for (int r = 0; r < 8; ++r) {
            int m = r + hi * 8;
            out[(size_t)(bb * SEQ + l0 + m) * INW + ct * 16 + n] = acc[tt][r];
        }
    }
}

extern "C" void kernel_launch(void* const* d_in, const int* in_sizes, int n_in,
                              void* d_out, int out_size, void* d_ws, size_t ws_size,
                              hipStream_t stream) {
    const float* x    = (const float*)d_in[0];
    const float* pos  = (const float*)d_in[1];
    const float* Wuv  = (const float*)d_in[2];
    const float* Wout = (const float*)d_in[3];

    _Float16* ws = (_Float16*)d_ws;
    const size_t nUV = (size_t)BATCH * SEQ * HIDDEN;   // 12,582,912
    const size_t nQK = (size_t)BATCH * SEQ * KEY;      //  2,097,152
    _Float16* U  = ws;
    _Float16* V  = U + nUV;
    _Float16* Q  = V + nUV;
    _Float16* Kb = Q + nQK;
    _Float16* G  = Kb + nQK;                           // total ~80 MB of ws

    const int row_tiles = BATCH * SEQ / 16;            // 1024

    proj_kernel<<<dim3(row_tiles, UVQK / 128), 256, 0, stream>>>(x, pos, Wuv, U, V, Q, Kb);
    attn_kernel<<<row_tiles, 256, LDS_TOTAL, stream>>>(Q, Kb, V, U, G);
    outgemm_kernel<<<row_tiles, 256, 0, stream>>>(G, Wout, (float*)d_out);
}
